// Model_60198261620950
// MI455X (gfx1250) — matrix operand order repaired and checked by reading, not compiled, not run on hardware
//
#include <hip/hip_runtime.h>

// ---------------------------------------------------------------------------
// e3nn-style equivariant GNN for MI455X (gfx1250, wave32, WMMA).
// - Node linears: f16 WMMA GEMMs (software-pipelined, double-buffered frags)
//   against densified block-sparse weights stored in B-fragment order.
// - Node features kept in f16 with padded stride 128 (free K-padding, no cvt
//   on the A path); only the atomically-accumulated agg stays f32.
// - Edge tensor products scatter with native f32 global atomics into an
//   L2-resident (33 MB << 192 MB L2) aggregation buffer.
// ---------------------------------------------------------------------------

typedef __attribute__((ext_vector_type(16))) _Float16 v16h;
typedef __attribute__((ext_vector_type(8)))  float    v8f;

#define DEV __device__ __forceinline__

static constexpr int NN = 20000;   // nodes
static constexpr int NE = 320000;  // edges

struct Params { const float* p[106]; };

DEV void aadd(float* p, float v) { unsafeAtomicAdd(p, v); }

DEV float sigmoidf_(float x) { return 1.f / (1.f + __expf(-x)); }
DEV float geluf_(float x) {
  const float k = 0.7978845608028654f;
  return 0.5f * x * (1.f + tanhf(k * (x + 0.044715f * x * x * x)));
}

// real CG 1 (x) 2 -> 1, unit Frobenius norm; l=1 order (y,z,x),
// l=2 order (xy, yz, z2, xz, x2-y2).
DEV void cg121(float hy, float hz, float hx, const float* s, float* o) {
  const float r10 = 0.31622776601683794f, r3 = 0.5773502691896258f;
  float sxy = s[0], syz = s[1], sz2 = s[2], sxz = s[3], sx2 = s[4];
  o[0] = (sxy * hx - sx2 * hy - r3 * sz2 * hy + syz * hz) * r10;
  o[1] = (sxz * hx + syz * hy + 2.f * r3 * sz2 * hz) * r10;
  o[2] = (sx2 * hx - r3 * sz2 * hx + sxy * hy + sxz * hz) * r10;
}

// ------------------------- densified weight entries -------------------------
// lin1: GATE_OUT(112) -> GATE_OUT(112), block diag, /sqrt(fan)
DEV float w1big_entry(const Params& P, int b, int k, int c) {
  const float r32 = 0.17677669529663687f, r8 = 0.35355339059327373f;
  if (k < 32)  { if (c < 32)              return P.p[b+0][k*32 + c] * r32; return 0.f; }
  if (k < 64)  { if (c >= 32 && c < 64)   return P.p[b+1][(k-32)*32 + (c-32)] * r32; return 0.f; }
  if (k < 88)  { if (c >= 64 && c < 88) { int u=(k-64)/3, m=(k-64)%3, v=(c-64)/3;
                   if (m == (c-64)%3)     return P.p[b+2][u*8 + v] * r8; } return 0.f; }
  if (k < 112) { if (c >= 88 && c < 112){ int u=(k-88)/3, m=(k-88)%3, v=(c-88)/3;
                   if (m == (c-88)%3)     return P.p[b+3][u*8 + v] * r8; } return 0.f; }
  return 0.f;
}
// sc: GATE_OUT(112) -> GATE_IN(128), /sqrt(fan)
DEV float wsc_entry(const Params& P, int b, int k, int c) {
  const float r32 = 0.17677669529663687f, r8 = 0.35355339059327373f;
  if (c < 32)  { if (k < 32)            return P.p[b+0][k*32 + c] * r32; return 0.f; }
  if (c < 64)  { if (k >= 32 && k < 64) return P.p[b+1][(k-32)*32 + (c-32)] * r32; return 0.f; }
  if (c < 72)  { if (k < 32)            return P.p[b+2][k*8 + (c-64)] * r32; return 0.f; }
  if (c < 80)  { if (k < 32)            return P.p[b+3][k*8 + (c-72)] * r32; return 0.f; }
  if (c < 104) { if (k >= 64 && k < 88) { int u=(k-64)/3, m=(k-64)%3, v=(c-80)/3;
                   if (m == (c-80)%3)   return P.p[b+4][u*8 + v] * r8; } return 0.f; }
  if (k >= 88 && k < 112) { int u=(k-88)/3, m=(k-88)%3, v=(c-104)/3;
                   if (m == (c-104)%3)  return P.p[b+5][u*8 + v] * r8; }
  return 0.f;
}
// lin2: mid(416) -> GATE_IN(128); folds /sqrt(fan) and /sqrt(1.5)
DEV float w2big_entry(const Params& P, int b, int k, int c) {
  const float s40 = 0.12909944487358055f;  // 1/sqrt(40*1.5)
  const float s56 = 0.10910894511799618f;  // 1/sqrt(56*1.5)
  if (c < 32) {
    if (k < 32)              return P.p[b+0][k*32 + c] * s40;
    if (k >= 360 && k < 368) return P.p[b+0][(32 + k-360)*32 + c] * s40;
    return 0.f;
  }
  if (c < 64) { int v = c-32;
    if (k >= 128 && k < 160) return P.p[b+1][(k-128)*32 + v] * s40;
    if (k >= 280 && k < 288) return P.p[b+1][(32 + k-280)*32 + v] * s40;
    return 0.f;
  }
  if (c < 72) { int v = c-64;
    if (k < 32)              return P.p[b+2][k*8 + v] * s40;
    if (k >= 360 && k < 368) return P.p[b+2][(32 + k-360)*8 + v] * s40;
    return 0.f;
  }
  if (c < 80) { int v = c-72;
    if (k < 32)              return P.p[b+3][k*8 + v] * s40;
    if (k >= 360 && k < 368) return P.p[b+3][(32 + k-360)*8 + v] * s40;
    return 0.f;
  }
  if (c < 104) { int j = c-80, v = j/3, m = j%3;
    if (k >= 160 && k < 256) { int kk=k-160; if (kk%3==m) return P.p[b+4][(kk/3)*8 + v] * s56; return 0.f; }
    if (k >= 256 && k < 280) { int kk=k-256; if (kk%3==m) return P.p[b+4][(32+kk/3)*8 + v] * s56; return 0.f; }
    if (k >= 312 && k < 336) { int kk=k-312; if (kk%3==m) return P.p[b+4][(40+kk/3)*8 + v] * s56; return 0.f; }
    if (k >= 368 && k < 392) { int kk=k-368; if (kk%3==m) return P.p[b+4][(48+kk/3)*8 + v] * s56; return 0.f; }
    return 0.f;
  }
  { int j = c-104, v = j/3, m = j%3;
    if (k >= 32  && k < 128) { int kk=k-32;  if (kk%3==m) return P.p[b+5][(kk/3)*8 + v] * s56; return 0.f; }
    if (k >= 288 && k < 312) { int kk=k-288; if (kk%3==m) return P.p[b+5][(32+kk/3)*8 + v] * s56; return 0.f; }
    if (k >= 336 && k < 360) { int kk=k-336; if (kk%3==m) return P.p[b+5][(40+kk/3)*8 + v] * s56; return 0.f; }
    if (k >= 392 && k < 416) { int kk=k-392; if (kk%3==m) return P.p[b+5][(48+kk/3)*8 + v] * s56; return 0.f; }
  }
  return 0.f;
}

// Build f16 weights directly in WMMA B-fragment order:
// dst[((tile*Ksteps + s)*32 + lane)*16 + j] = W[s*32 + (lane>=16?8:0) + j + (j>=8?8:0)][tile*16 + (lane&15)]
__global__ void build_frag(Params P, int b, int kind, _Float16* dst,
                           int Kdim, int Ksteps, int Ntiles) {
  int idx = blockIdx.x * blockDim.x + threadIdx.x;
  int total = Ntiles * Ksteps * 512;
  if (idx >= total) return;
  int j = idx & 15, l = (idx >> 4) & 31;
  int ts = idx >> 9, s = ts % Ksteps, t = ts / Ksteps;
  int k = s*32 + ((l & 16) ? 8 : 0) + j + ((j & 8) ? 8 : 0);
  int c = t*16 + (l & 15);
  float w = 0.f;
  if (k < Kdim) {
    if (kind == 0)      w = w1big_entry(P, b, k, c);
    else if (kind == 1) w = wsc_entry(P, b, k, c);
    else                w = w2big_entry(P, b, k, c);
  }
  dst[idx] = (_Float16)w;
}

// ------------------------------ WMMA helpers -------------------------------
// A-fragment (16x32 f16) from f32 row-major matrix (cvt path, used for agg).
DEV v16h load_a_frag_f32(const float* __restrict__ X, int row, int stride, int kb) {
  int l = threadIdx.x & 31;
  int lo = (l & 16) ? 8 : 0;
  v16h a;
#pragma unroll
  for (int v = 0; v < 8; ++v) {
    int ko = ((v & 3) << 1) + ((v >> 2) << 4);
    int k = kb + lo + ko;
    a[2*v]   = (_Float16)X[row*stride + k];
    a[2*v+1] = (_Float16)X[row*stride + k + 1];
  }
  return a;
}
// A-fragment from f16 matrix with padded row stride 128 (pure dword loads).
DEV v16h load_a_frag_h(const _Float16* __restrict__ X, int row, int kb) {
  int l = threadIdx.x & 31;
  int lo = (l & 16) ? 8 : 0;
  const unsigned int* r = (const unsigned int*)(X + (size_t)row * 128);
  union { unsigned int u[8]; v16h h; } cv;
#pragma unroll
  for (int v = 0; v < 8; ++v) {
    int ko = ((v & 3) << 1) + ((v >> 2) << 4);
    cv.u[v] = r[(kb + lo + ko) >> 1];
  }
  return cv.h;
}

// h = feat16 @ W1big : K=112 (4 ksteps, zero-padded rows), Ncols=112 (7 tiles)
__global__ void __launch_bounds__(64)
gemm_lin1(const _Float16* __restrict__ feat, float* __restrict__ h,
          const _Float16* __restrict__ wf, int n) {
  int wave = blockIdx.x * 2 + (threadIdx.x >> 5);
  int node0 = wave * 16;
  if (node0 >= n) return;
  int l = threadIdx.x & 31;
  int row = node0 + (l & 15);
  v8f zero = {0.f,0.f,0.f,0.f,0.f,0.f,0.f,0.f};
  v8f acc[7];
#pragma unroll
  for (int t = 0; t < 7; ++t) acc[t] = zero;
  // software pipeline: stage s+1 loads before s's WMMAs
  v16h a_cur = load_a_frag_h(feat, row, 0);
  v16h b_cur[7];
#pragma unroll
  for (int t = 0; t < 7; ++t) b_cur[t] = *(const v16h*)(wf + ((t*4)*32 + l)*16);
  for (int s = 0; s < 4; ++s) {
    v16h a_nxt; v16h b_nxt[7];
    if (s + 1 < 4) {
      a_nxt = load_a_frag_h(feat, row, (s+1)*32);
#pragma unroll
      for (int t = 0; t < 7; ++t)
        b_nxt[t] = *(const v16h*)(wf + ((t*4 + s + 1)*32 + l)*16);
    }
#pragma unroll
    for (int t = 0; t < 7; ++t)
      acc[t] = __builtin_amdgcn_wmma_f32_16x16x32_f16(false, a_cur, false, b_cur[t],
                                                      (short)0, acc[t], false, false);
    if (s + 1 < 4) {
      a_cur = a_nxt;
#pragma unroll
      for (int t = 0; t < 7; ++t) b_cur[t] = b_nxt[t];
    }
  }
  int Mo = node0 + ((l >> 4) << 3), c0 = l & 15;
#pragma unroll
  for (int t = 0; t < 7; ++t)
#pragma unroll
    for (int r = 0; r < 8; ++r)
      h[(Mo + r)*112 + t*16 + c0] = acc[t][r];
}

// out = agg @ W2big + feat16 @ WscBig, then gate -> feat16 (in place per tile)
__global__ void __launch_bounds__(64)
gemm_lin2_gate(const float* __restrict__ agg, _Float16* __restrict__ feat,
               const _Float16* __restrict__ w2f, const _Float16* __restrict__ wsf,
               int n) {
  __shared__ float lds[2][16][128];
  int ws_ = threadIdx.x >> 5;
  int wave = blockIdx.x * 2 + ws_;
  int node0 = wave * 16;
  bool active = (node0 < n);
  int l = threadIdx.x & 31;
  if (active) {
    int row = node0 + (l & 15);
    v8f zero = {0.f,0.f,0.f,0.f,0.f,0.f,0.f,0.f};
    v8f acc[8];
#pragma unroll
    for (int t = 0; t < 8; ++t) acc[t] = zero;

    // --- stage 1: agg (f32, K = 416 = 13*32), double-buffered ---
    v16h a_cur = load_a_frag_f32(agg, row, 416, 0);
    v16h b_cur[8];
#pragma unroll
    for (int t = 0; t < 8; ++t) b_cur[t] = *(const v16h*)(w2f + ((t*13)*32 + l)*16);
    for (int s = 0; s < 13; ++s) {
      v16h a_nxt; v16h b_nxt[8];
      if (s + 1 < 13) {
        a_nxt = load_a_frag_f32(agg, row, 416, (s+1)*32);
#pragma unroll
        for (int t = 0; t < 8; ++t)
          b_nxt[t] = *(const v16h*)(w2f + ((t*13 + s + 1)*32 + l)*16);
      }
#pragma unroll
      for (int t = 0; t < 8; ++t)
        acc[t] = __builtin_amdgcn_wmma_f32_16x16x32_f16(false, a_cur, false, b_cur[t],
                                                        (short)0, acc[t], false, false);
      if (s + 1 < 13) {
        a_cur = a_nxt;
#pragma unroll
        for (int t = 0; t < 8; ++t) b_cur[t] = b_nxt[t];
      }
    }

    // --- stage 2: skip connection (f16 feat, K = 112 padded to 128) ---
    a_cur = load_a_frag_h(feat, row, 0);
#pragma unroll
    for (int t = 0; t < 8; ++t) b_cur[t] = *(const v16h*)(wsf + ((t*4)*32 + l)*16);
    for (int s = 0; s < 4; ++s) {
      v16h a_nxt; v16h b_nxt[8];
      if (s + 1 < 4) {
        a_nxt = load_a_frag_h(feat, row, (s+1)*32);
#pragma unroll
        for (int t = 0; t < 8; ++t)
          b_nxt[t] = *(const v16h*)(wsf + ((t*4 + s + 1)*32 + l)*16);
      }
#pragma unroll
      for (int t = 0; t < 8; ++t)
        acc[t] = __builtin_amdgcn_wmma_f32_16x16x32_f16(false, a_cur, false, b_cur[t],
                                                        (short)0, acc[t], false, false);
      if (s + 1 < 4) {
        a_cur = a_nxt;
#pragma unroll
        for (int t = 0; t < 8; ++t) b_cur[t] = b_nxt[t];
      }
    }

    int Mo = (l >> 4) << 3, c0 = l & 15;
#pragma unroll
    for (int t = 0; t < 8; ++t)
#pragma unroll
      for (int r = 0; r < 8; ++r)
        lds[ws_][Mo + r][t*16 + c0] = acc[t][r];
  }
  __syncthreads();
  if (active) {
    for (int i = l; i < 16*128; i += 32) {
      int rr = i >> 7, c = i & 127;
      const float* o = lds[ws_][rr];
      float val = 0.f;                      // cols 112..127 are K-padding
      if (c < 32)       val = geluf_(o[c]);
      else if (c < 64)  val = tanhf(o[c]);
      else if (c < 88)  { int j = c-64; val = o[80 + j] * sigmoidf_(o[64 + j/3]); }
      else if (c < 112) { int j = c-88; val = o[104 + j] * sigmoidf_(o[72 + j/3]); }
      feat[(size_t)(node0 + rr)*128 + c] = (_Float16)val;
    }
  }
}

// ------------------------------- edge kernels ------------------------------
__global__ void edge_tp123(Params P, int tb, const float* __restrict__ h,
                           const int* __restrict__ esrc, const int* __restrict__ edst,
                           const float* __restrict__ sh, float* __restrict__ agg, int E) {
  int e = blockIdx.x * blockDim.x + threadIdx.x;
  if (e >= E) return;
  const float* s = sh + (size_t)e * 9;
  float s0 = s[0], v0 = s[1], v1 = s[2], v2 = s[3];
  float sd[5] = {s[4], s[5], s[6], s[7], s[8]};
  const float* hb = h + (size_t)esrc[e] * 112;
  __builtin_prefetch(hb, 0, 0);
  float* ar = agg + (size_t)edst[e] * 416;
  const float r3 = 0.5773502691896258f, r6 = 0.4082482904638630f;
  const float *w1 = P.p[tb+0], *w2 = P.p[tb+1], *w3 = P.p[tb+2], *w4 = P.p[tb+3];
  const float *w5 = P.p[tb+4], *w6 = P.p[tb+5], *w7 = P.p[tb+6], *w8 = P.p[tb+7];
  const float *w9 = P.p[tb+8], *wa = P.p[tb+9], *wb = P.p[tb+10], *wc = P.p[tb+11];
  for (int u = 0; u < 32; ++u) {              // paths 1,2 (even scalars)
    float hu = hb[u];
    aadd(ar + u, hu * s0 * w1[u]);
    float c = hu * w2[u] * r3;
    aadd(ar+32+3*u+0, c*v0); aadd(ar+32+3*u+1, c*v1); aadd(ar+32+3*u+2, c*v2);
  }
  for (int u = 0; u < 32; ++u) {              // paths 3,4 (odd scalars)
    float hu = hb[32 + u];
    aadd(ar + 128 + u, hu * s0 * w3[u]);
    float c = hu * w4[u] * r3;
    aadd(ar+160+3*u+0, c*v0); aadd(ar+160+3*u+1, c*v1); aadd(ar+160+3*u+2, c*v2);
  }
  for (int u = 0; u < 8; ++u) {               // paths 5-8 (even vectors)
    float a0 = hb[64+3*u], a1 = hb[64+3*u+1], a2 = hb[64+3*u+2];
    float c5 = s0 * w5[u] * r3;
    aadd(ar+256+3*u+0, a0*c5); aadd(ar+256+3*u+1, a1*c5); aadd(ar+256+3*u+2, a2*c5);
    aadd(ar+280+u, (a0*v0 + a1*v1 + a2*v2) * w6[u] * r3);
    float k7 = w7[u] * r6;
    aadd(ar+288+3*u+0, (a1*v2 - a2*v1)*k7);
    aadd(ar+288+3*u+1, (a2*v0 - a0*v2)*k7);
    aadd(ar+288+3*u+2, (a0*v1 - a1*v0)*k7);
    float o[3]; cg121(a0, a1, a2, sd, o);
    aadd(ar+312+3*u+0, o[0]*w8[u]); aadd(ar+312+3*u+1, o[1]*w8[u]); aadd(ar+312+3*u+2, o[2]*w8[u]);
  }
  for (int u = 0; u < 8; ++u) {               // paths 9-12 (odd vectors)
    float a0 = hb[88+3*u], a1 = hb[88+3*u+1], a2 = hb[88+3*u+2];
    float c9 = s0 * w9[u] * r3;
    aadd(ar+336+3*u+0, a0*c9); aadd(ar+336+3*u+1, a1*c9); aadd(ar+336+3*u+2, a2*c9);
    aadd(ar+360+u, (a0*v0 + a1*v1 + a2*v2) * wa[u] * r3);
    float kk = wb[u] * r6;
    aadd(ar+368+3*u+0, (a1*v2 - a2*v1)*kk);
    aadd(ar+368+3*u+1, (a2*v0 - a0*v2)*kk);
    aadd(ar+368+3*u+2, (a0*v1 - a1*v0)*kk);
    float o[3]; cg121(a0, a1, a2, sd, o);
    aadd(ar+392+3*u+0, o[0]*wc[u]); aadd(ar+392+3*u+1, o[1]*wc[u]); aadd(ar+392+3*u+2, o[2]*wc[u]);
  }
}

__global__ void edge_tp0(Params P, const float* __restrict__ h,
                         const int* __restrict__ esrc, const int* __restrict__ edst,
                         const float* __restrict__ sh, float* __restrict__ agg, int E) {
  int e = blockIdx.x * blockDim.x + threadIdx.x;
  if (e >= E) return;
  const float* s = sh + (size_t)e * 9;
  float h0 = h[esrc[e]];
  float* ar = agg + (size_t)edst[e] * 4;
  const float r3 = 0.5773502691896258f;
  float w0 = P.p[8][0], w1 = P.p[9][0];
  aadd(ar + 0, h0 * s[0] * w0);
  float c = h0 * w1 * r3;
  aadd(ar+1, c*s[1]); aadd(ar+2, c*s[2]); aadd(ar+3, c*s[3]);
}

__global__ void edge_tp4(Params P, const float* __restrict__ h,
                         const int* __restrict__ esrc, const int* __restrict__ edst,
                         const float* __restrict__ sh, float* __restrict__ agg, int E) {
  int e = blockIdx.x * blockDim.x + threadIdx.x;
  if (e >= E) return;
  const float* s = sh + (size_t)e * 9;
  float s0 = s[0], v0 = s[1], v1 = s[2], v2 = s[3];
  const float* hb = h + (size_t)esrc[e] * 112;
  __builtin_prefetch(hb, 0, 0);
  float* ar = agg + (size_t)edst[e] * 80;
  const float r3 = 0.5773502691896258f;
  const float *w1 = P.p[102], *w2 = P.p[103], *w3 = P.p[104], *w4 = P.p[105];
  for (int u = 0; u < 32; ++u) aadd(ar + u,      hb[u]      * s0 * w1[u]);
  for (int u = 0; u < 32; ++u) aadd(ar + 32 + u, hb[32 + u] * s0 * w2[u]);
  for (int u = 0; u < 8; ++u) {
    float d = hb[64+3*u]*v0 + hb[64+3*u+1]*v1 + hb[64+3*u+2]*v2;
    aadd(ar + 64 + u, d * w3[u] * r3);
  }
  for (int u = 0; u < 8; ++u) {
    float d = hb[88+3*u]*v0 + hb[88+3*u+1]*v1 + hb[88+3*u+2]*v2;
    aadd(ar + 72 + u, d * w4[u] * r3);
  }
}

// ------------------------------- node kernels ------------------------------
__global__ void node0_pre(Params P, const float* __restrict__ x,
                          float* __restrict__ h, int n) {
  int i = blockIdx.x * blockDim.x + threadIdx.x;
  if (i < n) h[i] = x[i] * P.p[0][0];
}

__global__ void node0_post(Params P, const float* __restrict__ x,
                           const float* __restrict__ agg, _Float16* __restrict__ feat,
                           int n) {
  int i = blockIdx.x * blockDim.x + threadIdx.x;
  if (i >= n) return;
  const float inv15 = 0.8164965809277260f;   // 1/sqrt(1.5)
  float xv = x[i];
  float a0 = agg[i*4+0]*inv15, a1 = agg[i*4+1]*inv15,
        a2 = agg[i*4+2]*inv15, a3 = agg[i*4+3]*inv15;
  const float *L20 = P.p[1], *L23 = P.p[3], *L25 = P.p[4];
  const float *S0 = P.p[5], *S3 = P.p[7];
  _Float16* f = feat + (size_t)i * 128;
  for (int v = 0; v < 32; ++v) f[v] = (_Float16)geluf_(xv*S0[v] + a0*L20[v]);
  for (int v = 32; v < 64; ++v) f[v] = (_Float16)0.f;   // tanh(0)
  for (int j = 64; j < 88; ++j) f[j] = (_Float16)0.f;   // v1 block is zero
  float am[3] = {a1, a2, a3};
  for (int v = 0; v < 8; ++v) {
    float g = sigmoidf_(xv*S3[v] + a0*L23[v]);
    for (int m = 0; m < 3; ++m) f[88 + 3*v + m] = (_Float16)(am[m] * L25[v] * g);
  }
  for (int j = 112; j < 128; ++j) f[j] = (_Float16)0.f; // K-padding
}

__global__ void node4_post(Params P, const _Float16* __restrict__ feat,
                           const float* __restrict__ agg, float* __restrict__ out, int n) {
  int i = blockIdx.x * blockDim.x + threadIdx.x;
  if (i >= n) return;
  const float s60 = 0.1290994448735806f;   // 1/sqrt(40*1.5)
  const float s32 = 0.1767766952966369f;   // 1/sqrt(32)
  const float* a = agg + (size_t)i * 80;
  const _Float16* f = feat + (size_t)i * 128;
  const float *L20 = P.p[98], *L21 = P.p[99], *S0 = P.p[100], *S1 = P.p[101];
  float o0 = 0.f;
  for (int u = 0; u < 32; ++u) o0 += a[32 + u] * L20[u];
  for (int u = 0; u < 8;  ++u) o0 += a[64 + u] * L20[32 + u];
  o0 *= s60;
  float sc0 = 0.f;
  for (int u = 0; u < 32; ++u) sc0 += (float)f[32 + u] * S0[u];
  out[i*7 + 0] = o0 + sc0 * s32;
  for (int v = 0; v < 6; ++v) {
    float o = 0.f;
    for (int u = 0; u < 32; ++u) o += a[u] * L21[u*6 + v];
    for (int u = 0; u < 8;  ++u) o += a[72 + u] * L21[(32 + u)*6 + v];
    o *= s60;
    float sc = 0.f;
    for (int u = 0; u < 32; ++u) sc += (float)f[u] * S1[u*6 + v];
    out[i*7 + 1 + v] = o + sc * s32;
  }
}

// --------------------------------- launcher --------------------------------
extern "C" void kernel_launch(void* const* d_in, const int* in_sizes, int n_in,
                              void* d_out, int out_size, void* d_ws, size_t ws_size,
                              hipStream_t stream) {
  (void)in_sizes; (void)out_size; (void)ws_size;
  const float* x    = (const float*)d_in[0];
  const int*   esrc = (const int*)d_in[1];
  const int*   edst = (const int*)d_in[2];
  const float* sh   = (const float*)d_in[3];

  Params P;
  for (int i = 0; i < 106; ++i)
    P.p[i] = (4 + i < n_in) ? (const float*)d_in[4 + i] : nullptr;

  // workspace layout
  _Float16* feat = (_Float16*)d_ws;                         // NN*128 (padded)
  float* hbuf = (float*)(feat + (size_t)NN * 128);          // NN*112
  float* agg  = hbuf + (size_t)NN * 112;                    // NN*416
  _Float16* f16b = (_Float16*)(agg + (size_t)NN * 416);
  _Float16* w1f[4]; for (int i = 0; i < 4; ++i) w1f[i] = f16b + (size_t)i * 14336;
  _Float16* w2b = f16b + (size_t)4 * 14336;
  _Float16* w2f[3]; for (int i = 0; i < 3; ++i) w2f[i] = w2b + (size_t)i * 53248;
  _Float16* wsb = w2b + (size_t)3 * 53248;
  _Float16* wsf[3]; for (int i = 0; i < 3; ++i) wsf[i] = wsb + (size_t)i * 16384;

  const int nodeBlocks = (NN + 255) / 256;
  const int edgeBlocks = NE / 256;          // 320000 = 1250*256
  const int gemmBlocks = (NN / 16) / 2;     // 1250 wave-tiles, 2 waves/block

  // densify weights into WMMA fragment layout (leaf bases per sorted pytree)
  for (int L = 1; L <= 4; ++L) {
    int b = (L <= 3) ? (10 + (L - 1) * 28) : 94;
    build_frag<<<(7*4*512 + 255)/256, 256, 0, stream>>>(P, b, 0, w1f[L-1], 112, 4, 7);
  }
  for (int L = 1; L <= 3; ++L) {
    int cb = 10 + (L - 1) * 28;
    build_frag<<<(8*13*512 + 255)/256, 256, 0, stream>>>(P, cb + 4, 2, w2f[L-1], 416, 13, 8);
    build_frag<<<(8*4*512  + 255)/256, 256, 0, stream>>>(P, cb + 10, 1, wsf[L-1], 112, 4, 8);
  }

  // ---- layer 0 (scalar input, VALU) ----
  hipMemsetAsync(agg, 0, (size_t)NN * 4 * sizeof(float), stream);
  node0_pre<<<nodeBlocks, 256, 0, stream>>>(P, x, hbuf, NN);
  edge_tp0<<<edgeBlocks, 256, 0, stream>>>(P, hbuf, esrc, edst, sh, agg, NE);
  node0_post<<<nodeBlocks, 256, 0, stream>>>(P, x, agg, feat, NN);

  // ---- layers 1..3 (WMMA node linears + atomic edge TP) ----
  for (int L = 1; L <= 3; ++L) {
    hipMemsetAsync(agg, 0, (size_t)NN * 416 * sizeof(float), stream);
    gemm_lin1<<<gemmBlocks, 64, 0, stream>>>(feat, hbuf, w1f[L-1], NN);
    edge_tp123<<<edgeBlocks, 256, 0, stream>>>(P, 10 + (L-1)*28 + 16, hbuf,
                                               esrc, edst, sh, agg, NE);
    gemm_lin2_gate<<<gemmBlocks, 64, 0, stream>>>(agg, feat, w2f[L-1], wsf[L-1], NN);
  }

  // ---- layer 4 (7-dim output) ----
  hipMemsetAsync(agg, 0, (size_t)NN * 80 * sizeof(float), stream);
  gemm_lin1<<<gemmBlocks, 64, 0, stream>>>(feat, hbuf, w1f[3], NN);
  edge_tp4<<<edgeBlocks, 256, 0, stream>>>(P, hbuf, esrc, edst, sh, agg, NE);
  node4_post<<<nodeBlocks, 256, 0, stream>>>(P, feat, agg, (float*)d_out, NN);
}
